// DecoderPromptLayer_79989470920966
// MI455X (gfx1250) — matrix operand order repaired and checked by reading, not compiled, not run on hardware
//
#include <hip/hip_runtime.h>

// ---------------------------------------------------------------------------
// Problem constants (from reference)
// ---------------------------------------------------------------------------
static constexpr int cB  = 2;
static constexpr int cT  = 2048;
static constexpr int cM  = 256;
static constexpr int cD  = 1024;
static constexpr int cH  = 16;
static constexpr int cDH = 64;
static constexpr int cDF = 4096;
static constexpr int cS  = cM + cT;       // 2304
static constexpr int cBS = cB * cS;       // 4608

typedef __bf16 bf16;
typedef __bf16 v16bf __attribute__((ext_vector_type(16)));
typedef float  v8f   __attribute__((ext_vector_type(8)));
typedef int    gv4i  __attribute__((vector_size(16)));   // matches builtin param

// ---------------------------------------------------------------------------
// gfx1250 async global->LDS staging (ASYNCcnt-tracked). Falls back to
// synchronous vector copies if the builtin is unavailable on this toolchain.
// ---------------------------------------------------------------------------
__device__ inline void async_copy_b128(const void* g, void* l) {
#if defined(__has_builtin) && __has_builtin(__builtin_amdgcn_global_load_async_to_lds_b128)
    __builtin_amdgcn_global_load_async_to_lds_b128(
        (__attribute__((address_space(1))) gv4i*)g,
        (__attribute__((address_space(3))) gv4i*)l,
        /*offset=*/0, /*cpol=*/0);
#else
    *reinterpret_cast<uint4*>(l) = *reinterpret_cast<const uint4*>(g);
#endif
}

__device__ inline void wait_async() {
#if defined(__has_builtin) && __has_builtin(__builtin_amdgcn_s_wait_asynccnt)
    __builtin_amdgcn_s_wait_asynccnt(0);
#else
    asm volatile("s_wait_asynccnt 0" ::: "memory");
#endif
}

// ---------------------------------------------------------------------------
// WMMA fragment helpers (contiguous k = 16*(lane/16) + s, wave32)
// ---------------------------------------------------------------------------
__device__ inline v8f wmma_bf16(v16bf a, v16bf b, v8f c) {
    return __builtin_amdgcn_wmma_f32_16x16x32_bf16(
        /*neg_a=*/false, a, /*neg_b=*/false, b,
        /*c_mod=*/(short)0, c, /*reuse_a=*/false, /*reuse_b=*/false);
}

// A-matrix 16x32 (MxK), memory layout [m][k] with given element stride.
// lanes 0-15: M=lane, K=0..15; lanes 16-31: same M, K=16..31
__device__ inline v16bf frag_a_mk(const bf16* base, int stride) {
    const int lane = threadIdx.x & 31;
    const int m  = lane & 15;
    const int kh = (lane >> 4) * 16;
    v16bf r;
#pragma unroll
    for (int i = 0; i < 4; ++i) {
        r[2*i]     = base[m * stride + kh + 2*i];
        r[2*i+1]   = base[m * stride + kh + 2*i + 1];
        r[8+2*i]   = base[m * stride + 8 + kh + 2*i];
        r[8+2*i+1] = base[m * stride + 8 + kh + 2*i + 1];
    }
    return r;
}

// B-matrix 32x16 (KxN), memory layout [k][n] with given stride.
// lanes 0-15: N=lane, K=0..15 (2 per VGPR); lanes 16-31: same N, K=16..31
__device__ inline v16bf frag_b_kn(const bf16* base, int stride) {
    const int lane = threadIdx.x & 31;
    const int n  = lane & 15;
    const int kh = (lane >> 4) * 16;
    v16bf r;
#pragma unroll
    for (int i = 0; i < 8; ++i) {
        r[2*i]   = base[(kh + 2*i)     * stride + n];
        r[2*i+1] = base[(kh + 2*i + 1) * stride + n];
    }
    return r;
}

// B-matrix 32x16 (KxN), memory layout [n][k] (i.e. transposed in memory).
__device__ inline v16bf frag_b_nk(const bf16* base, int stride) {
    const int lane = threadIdx.x & 31;
    const int n  = lane & 15;
    const int kh = (lane >> 4) * 16;
    v16bf r;
#pragma unroll
    for (int i = 0; i < 8; ++i) {
        r[2*i]   = base[n * stride + kh + 2*i];
        r[2*i+1] = base[n * stride + kh + 2*i + 1];
    }
    return r;
}

// ---------------------------------------------------------------------------
// fp32 -> bf16 conversion
// ---------------------------------------------------------------------------
__global__ void cvt_bf16_kernel(const float* __restrict__ in,
                                bf16* __restrict__ out, size_t n) {
    size_t i = (size_t)blockIdx.x * blockDim.x + threadIdx.x;
    if (i < n) out[i] = (bf16)in[i];
}

// ---------------------------------------------------------------------------
// LayerNorm (optionally fused with the memory/x concat).
// One block per row of 1024 floats; 256 threads x 4 elements.
// ---------------------------------------------------------------------------
__global__ void __launch_bounds__(256)
ln_kernel(const float* __restrict__ srcA,   // concat: memory [B,M,D]; else full [B,S,D]
          const float* __restrict__ srcB,   // concat: x [B,T,D]; else unused
          const float* __restrict__ g, const float* __restrict__ beta,
          float* __restrict__ xfull,        // optional fp32 copy of the input row
          bf16* __restrict__ y,             // normalized output, bf16
          int concat)
{
    __shared__ float s1[256];
    __shared__ float s2[256];
    const int row = blockIdx.x;            // 0 .. B*S-1
    const int b = row / cS;
    const int s = row % cS;
    const float* src;
    if (concat) {
        src = (s < cM) ? (srcA + ((size_t)b * cM + s) * cD)
                       : (srcB + ((size_t)b * cT + (s - cM)) * cD);
    } else {
        src = srcA + (size_t)row * cD;
    }
    const int t = threadIdx.x;
    float4 v = reinterpret_cast<const float4*>(src)[t];
    float sum = v.x + v.y + v.z + v.w;
    float sq  = v.x*v.x + v.y*v.y + v.z*v.z + v.w*v.w;
    s1[t] = sum; s2[t] = sq;
    __syncthreads();
#pragma unroll
    for (int st = 128; st > 0; st >>= 1) {
        if (t < st) { s1[t] += s1[t + st]; s2[t] += s2[t + st]; }
        __syncthreads();
    }
    const float mu   = s1[0] * (1.0f / cD);
    const float var  = s2[0] * (1.0f / cD) - mu * mu;
    const float rstd = rsqrtf(var + 1e-5f);

    if (xfull) reinterpret_cast<float4*>(xfull + (size_t)row * cD)[t] = v;

    const int base = t * 4;
    float e[4] = {v.x, v.y, v.z, v.w};
#pragma unroll
    for (int i = 0; i < 4; ++i) {
        float yn = (e[i] - mu) * rstd * g[base + i] + beta[base + i];
        y[(size_t)row * cD + base + i] = (bf16)yn;
    }
}

// ---------------------------------------------------------------------------
// Generic bf16 WMMA GEMM: C[M,N] = A[M,K] * W[K,N] (+ epilogue)
// Block: 256 threads (8 waves), 128x128 tile, BK=32, double-buffered LDS
// staged with gfx1250 async global->LDS copies.
// Wave (wm = wave>>1 in 0..3, wn = wave&1): 32x64 output = 2x4 fragments.
// EPI 0: fp32 out = acc + bias
// EPI 1: bf16 out = acc + bias
// EPI 2: fp32 out = acc + bias + resid
// EPI 3: bf16 out = relu(acc + bias)^2
// ---------------------------------------------------------------------------
template <int EPI>
__global__ void __launch_bounds__(256)
gemm_bf16_kernel(const bf16* __restrict__ A, int lda,
                 const bf16* __restrict__ W, int ldb,
                 const float* __restrict__ bias,
                 const float* __restrict__ resid,
                 float* __restrict__ outF,
                 bf16*  __restrict__ outB,
                 int K, int N)
{
    __shared__ bf16 As[2][128][40];    // 128x32 tile, pad 40
    __shared__ bf16 Bs[2][32][136];    // 32x128 tile, pad 136

    const int tid  = threadIdx.x;
    const int wave = tid >> 5;
    const int wm   = wave >> 1;            // 0..3
    const int wn   = wave & 1;             // 0..1
    const int row0 = blockIdx.y * 128;
    const int col0 = blockIdx.x * 128;

    v8f zero = {};
    v8f acc[2][4];
#pragma unroll
    for (int fm = 0; fm < 2; ++fm)
#pragma unroll
        for (int fn = 0; fn < 4; ++fn) acc[fm][fn] = zero;

    auto stage = [&](int buf, int kt) {
        // A tile: 128x32 bf16 = 512 x b128, 2 per thread
#pragma unroll
        for (int i = 0; i < 2; ++i) {
            int vi = tid + i * 256;
            int r  = vi >> 2;
            int cv = vi & 3;
            async_copy_b128(A + (size_t)(row0 + r) * lda + kt * 32 + cv * 8,
                            &As[buf][r][cv * 8]);
        }
        // B tile: 32x128 bf16 = 512 x b128, 2 per thread
#pragma unroll
        for (int i = 0; i < 2; ++i) {
            int vi = tid + i * 256;
            int r  = vi >> 4;
            int cv = vi & 15;
            async_copy_b128(W + (size_t)(kt * 32 + r) * ldb + col0 + cv * 8,
                            &Bs[buf][r][cv * 8]);
        }
    };

    const int kIters = K >> 5;
    stage(0, 0);
    for (int kt = 0; kt < kIters; ++kt) {
        const int buf = kt & 1;
        wait_async();
        __syncthreads();
        if (kt + 1 < kIters) stage(buf ^ 1, kt + 1);

        v16bf af[2], bfg[4];
        af[0] = frag_a_mk(&As[buf][wm * 32][0],      40);
        af[1] = frag_a_mk(&As[buf][wm * 32 + 16][0], 40);
#pragma unroll
        for (int fn = 0; fn < 4; ++fn)
            bfg[fn] = frag_b_kn(&Bs[buf][0][wn * 64 + fn * 16], 136);
#pragma unroll
        for (int fm = 0; fm < 2; ++fm)
#pragma unroll
            for (int fn = 0; fn < 4; ++fn)
                acc[fm][fn] = wmma_bf16(af[fm], bfg[fn], acc[fm][fn]);
    }

    // Epilogue
    const int lane = tid & 31;
    const int n  = lane & 15;
    const int mb = (lane >> 4) * 8;
#pragma unroll
    for (int fm = 0; fm < 2; ++fm) {
#pragma unroll
        for (int fn = 0; fn < 4; ++fn) {
#pragma unroll
            for (int r = 0; r < 8; ++r) {
                int row = row0 + wm * 32 + fm * 16 + mb + r;
                int col = col0 + wn * 64 + fn * 16 + n;
                size_t idx = (size_t)row * N + col;
                float v = acc[fm][fn][r] + bias[col];
                if (EPI == 0) {
                    outF[idx] = v;
                } else if (EPI == 1) {
                    outB[idx] = (bf16)v;
                } else if (EPI == 2) {
                    outF[idx] = v + resid[idx];
                } else { // EPI == 3: squared ReLU
                    v = v > 0.0f ? v : 0.0f;
                    outB[idx] = (bf16)(v * v);
                }
            }
        }
    }
}

// ---------------------------------------------------------------------------
// RoPE on Q/K (first R=32 of each 64-dim head), fp32 in -> bf16 out, [B,S,D]
// ---------------------------------------------------------------------------
__global__ void rope_kernel(const float* __restrict__ qf, const float* __restrict__ kf,
                            bf16* __restrict__ qo, bf16* __restrict__ ko)
{
    size_t idx = (size_t)blockIdx.x * blockDim.x + threadIdx.x;
    if (idx >= (size_t)cBS * cD) return;
    int d = (int)(idx % cD);
    size_t row = idx / cD;
    int s = (int)(row % cS);
    int dh = d % cDH;
    float qv = qf[idx], kv = kf[idx];
    if (dh < 32) {
        int d16 = dh & 15;
        float invf = __powf(10000.0f, -(float)d16 * (1.0f / 16.0f));
        float ang = (float)s * invf;
        float c = __cosf(ang), sn = __sinf(ang);
        if (dh < 16) {
            qv = qv * c - qf[idx + 16] * sn;
            kv = kv * c - kf[idx + 16] * sn;
        } else {
            qv = qv * c + qf[idx - 16] * sn;
            kv = kv * c + kf[idx - 16] * sn;
        }
    }
    qo[idx] = (bf16)qv;
    ko[idx] = (bf16)kv;
}

// ---------------------------------------------------------------------------
// Flash-style attention. Grid: (S/64, H, B). Block: 128 threads (4 waves).
// Wave w owns q-rows [qb + 16w, qb + 16w + 16). K/V tiles of 64 keys staged
// to LDS with async copies. Mask: allowed iff (j < M) || (j <= i).
// ---------------------------------------------------------------------------
__global__ void __launch_bounds__(128)
attn_kernel(const bf16* __restrict__ q, const bf16* __restrict__ k,
            const bf16* __restrict__ v, bf16* __restrict__ o)
{
    __shared__ bf16 Ks[64][72];
    __shared__ bf16 Vs[64][72];
    __shared__ bf16 Ps[4][16][72];

    const int tid  = threadIdx.x;
    const int w    = tid >> 5;
    const int lane = tid & 31;
    const int n    = lane & 15;
    const int mb   = (lane >> 4) * 8;

    const int qb = blockIdx.x * 64;
    const int h  = blockIdx.y;
    const int b  = blockIdx.z;

    const bf16* qbase = q + ((size_t)b * cS + qb + w * 16) * cD + h * cDH;
    v16bf qa0 = frag_a_mk(qbase,      cD);   // dh 0..31
    v16bf qa1 = frag_a_mk(qbase + 32, cD);   // dh 32..63

    v8f zero = {};
    v8f of[4];
    float rmax[8], rsum[8];
#pragma unroll
    for (int ct = 0; ct < 4; ++ct) of[ct] = zero;
#pragma unroll
    for (int r = 0; r < 8; ++r) { rmax[r] = -1e30f; rsum[r] = 0.0f; }

    const int iMaxRow = qb + 63;
    const int jmax = (iMaxRow >= cM) ? iMaxRow : (cM - 1);
    const int nkt  = (jmax >> 6) + 1;

    const bf16* kbase = k + (size_t)b * cS * cD + h * cDH;
    const bf16* vbase = v + (size_t)b * cS * cD + h * cDH;

    for (int kt = 0; kt < nkt; ++kt) {
        const int kb0 = kt * 64;
        __syncthreads();
        // stage K,V 64x64 tiles: 512 b128 each, 4 per thread, async to LDS
#pragma unroll
        for (int i = 0; i < 4; ++i) {
            int vi = tid + i * 128;
            int r  = vi >> 3;
            int cv = vi & 7;
            async_copy_b128(kbase + (size_t)(kb0 + r) * cD + cv * 8, &Ks[r][cv * 8]);
            async_copy_b128(vbase + (size_t)(kb0 + r) * cD + cv * 8, &Vs[r][cv * 8]);
        }
        wait_async();
        __syncthreads();

        // scores: 16 q-rows x 64 keys, 4 col-tiles
        float sv[4][8];
#pragma unroll
        for (int ct = 0; ct < 4; ++ct) {
            v8f sf = zero;
            v16bf bk0 = frag_b_nk(&Ks[ct * 16][0],  72);
            v16bf bk1 = frag_b_nk(&Ks[ct * 16][32], 72);
            sf = wmma_bf16(qa0, bk0, sf);
            sf = wmma_bf16(qa1, bk1, sf);
#pragma unroll
            for (int r = 0; r < 8; ++r) {
                float sc = sf[r] * 0.125f;         // 1/sqrt(64)
                int j = kb0 + ct * 16 + n;
                int i_abs = qb + w * 16 + mb + r;
                if (!((j < cM) || (j <= i_abs))) sc = -1e30f;
                sv[ct][r] = sc;
            }
        }

        // online softmax per row (rows live across 16-lane halves)
#pragma unroll
        for (int r = 0; r < 8; ++r) {
            float mx = sv[0][r];
#pragma unroll
            for (int ct = 1; ct < 4; ++ct) mx = fmaxf(mx, sv[ct][r]);
            mx = fmaxf(mx, __shfl_xor(mx, 1));
            mx = fmaxf(mx, __shfl_xor(mx, 2));
            mx = fmaxf(mx, __shfl_xor(mx, 4));
            mx = fmaxf(mx, __shfl_xor(mx, 8));
            float nm   = fmaxf(rmax[r], mx);
            float corr = __expf(rmax[r] - nm);
            rmax[r] = nm;
            float ls = 0.0f;
#pragma unroll
            for (int ct = 0; ct < 4; ++ct) {
                float p = __expf(sv[ct][r] - nm);
                sv[ct][r] = p;
                ls += p;
            }
            ls += __shfl_xor(ls, 1);
            ls += __shfl_xor(ls, 2);
            ls += __shfl_xor(ls, 4);
            ls += __shfl_xor(ls, 8);
            rsum[r] = rsum[r] * corr + ls;
#pragma unroll
            for (int ct = 0; ct < 4; ++ct) of[ct][r] *= corr;
            // stage P into per-wave LDS scratch (A-fragment re-layout)
#pragma unroll
            for (int ct = 0; ct < 4; ++ct)
                Ps[w][mb + r][ct * 16 + n] = (bf16)sv[ct][r];
        }
        // Same-wave DS ops are in-order; only the counter wait is needed.
        asm volatile("s_wait_dscnt 0" ::: "memory");

        // O += P @ V
        v16bf pa0 = frag_a_mk(&Ps[w][0][0],  72);
        v16bf pa1 = frag_a_mk(&Ps[w][0][32], 72);
#pragma unroll
        for (int ct = 0; ct < 4; ++ct) {
            v16bf vb0 = frag_b_kn(&Vs[0][ct * 16],  72);
            v16bf vb1 = frag_b_kn(&Vs[32][ct * 16], 72);
            of[ct] = wmma_bf16(pa0, vb0, of[ct]);
            of[ct] = wmma_bf16(pa1, vb1, of[ct]);
        }
    }

    // normalize and write o in [B,S,D] bf16
    bf16* obase = o + ((size_t)b * cS + qb + w * 16) * cD + h * cDH;
#pragma unroll
    for (int ct = 0; ct < 4; ++ct) {
#pragma unroll
        for (int r = 0; r < 8; ++r) {
            float val = of[ct][r] / rsum[r];
            obase[(size_t)(mb + r) * cD + ct * 16 + n] = (bf16)val;
        }
    }
}

// ---------------------------------------------------------------------------
// Extract x_full[:, M:] -> d_out (fp32)
// ---------------------------------------------------------------------------
__global__ void extract_kernel(const float* __restrict__ xf3, float* __restrict__ out)
{
    size_t i = (size_t)blockIdx.x * blockDim.x + threadIdx.x;
    if (i >= (size_t)cB * cT * cD) return;
    int d = (int)(i % cD);
    size_t bt = i / cD;
    int t = (int)(bt % cT);
    int b = (int)(bt / cT);
    out[i] = xf3[((size_t)b * cS + cM + t) * cD + d];
}

// ---------------------------------------------------------------------------
// Host orchestration
// ---------------------------------------------------------------------------
extern "C" void kernel_launch(void* const* d_in, const int* in_sizes, int n_in,
                              void* d_out, int out_size, void* d_ws, size_t ws_size,
                              hipStream_t stream) {
    (void)in_sizes; (void)n_in; (void)out_size; (void)ws_size;
    const float* x      = (const float*)d_in[0];
    const float* memory = (const float*)d_in[1];
    const float* Wq = (const float*)d_in[2];
    const float* bq = (const float*)d_in[3];
    const float* Wk = (const float*)d_in[4];
    const float* bk = (const float*)d_in[5];
    const float* Wv = (const float*)d_in[6];
    const float* bv = (const float*)d_in[7];
    const float* Wo = (const float*)d_in[8];
    const float* bo = (const float*)d_in[9];
    const float* W1 = (const float*)d_in[10];
    const float* b1 = (const float*)d_in[11];
    const float* W2 = (const float*)d_in[12];
    const float* b2 = (const float*)d_in[13];
    const float* ln1_g = (const float*)d_in[14];
    const float* ln1_b = (const float*)d_in[15];
    const float* ln2_g = (const float*)d_in[16];
    const float* ln2_b = (const float*)d_in[17];

    char* p = (char*)d_ws;
    auto take = [&](size_t bytes) -> char* {
        char* r = p;
        p += (bytes + 255) & ~(size_t)255;
        return r;
    };
    const size_t actF = (size_t)cBS * cD * sizeof(float);
    const size_t actB = (size_t)cBS * cD * sizeof(bf16);

    float* xfull  = (float*)take(actF);
    float* xfull2 = (float*)take(actF);
    float* xfull3 = (float*)take(actF);
    float* qf     = (float*)take(actF);
    float* kf     = (float*)take(actF);
    bf16* xa  = (bf16*)take(actB);
    bf16* qbb = (bf16*)take(actB);
    bf16* kbb = (bf16*)take(actB);
    bf16* vbb = (bf16*)take(actB);
    bf16* ob  = (bf16*)take(actB);
    bf16* xfb = (bf16*)take(actB);
    bf16* hb  = (bf16*)take((size_t)cBS * cDF * sizeof(bf16));
    bf16* wqB = (bf16*)take((size_t)cD * cD * sizeof(bf16));
    bf16* wkB = (bf16*)take((size_t)cD * cD * sizeof(bf16));
    bf16* wvB = (bf16*)take((size_t)cD * cD * sizeof(bf16));
    bf16* woB = (bf16*)take((size_t)cD * cD * sizeof(bf16));
    bf16* w1B = (bf16*)take((size_t)cD * cDF * sizeof(bf16));
    bf16* w2B = (bf16*)take((size_t)cDF * cD * sizeof(bf16));

    const size_t nW  = (size_t)cD * cD;
    const size_t nW1 = (size_t)cD * cDF;
    cvt_bf16_kernel<<<(unsigned)((nW  + 255) / 256), 256, 0, stream>>>(Wq, wqB, nW);
    cvt_bf16_kernel<<<(unsigned)((nW  + 255) / 256), 256, 0, stream>>>(Wk, wkB, nW);
    cvt_bf16_kernel<<<(unsigned)((nW  + 255) / 256), 256, 0, stream>>>(Wv, wvB, nW);
    cvt_bf16_kernel<<<(unsigned)((nW  + 255) / 256), 256, 0, stream>>>(Wo, woB, nW);
    cvt_bf16_kernel<<<(unsigned)((nW1 + 255) / 256), 256, 0, stream>>>(W1, w1B, nW1);
    cvt_bf16_kernel<<<(unsigned)((nW1 + 255) / 256), 256, 0, stream>>>(W2, w2B, nW1);

    // LN1 fused with concat(memory, x) -> xfull (fp32) + xa (bf16)
    ln_kernel<<<cBS, 256, 0, stream>>>(memory, x, ln1_g, ln1_b, xfull, xa, 1);

    // QKV projections (128x128 tiles)
    dim3 gD(cD / 128, cBS / 128);
    gemm_bf16_kernel<0><<<gD, 256, 0, stream>>>(xa, cD, wqB, cD, bq, nullptr, qf, nullptr, cD, cD);
    gemm_bf16_kernel<0><<<gD, 256, 0, stream>>>(xa, cD, wkB, cD, bk, nullptr, kf, nullptr, cD, cD);
    gemm_bf16_kernel<1><<<gD, 256, 0, stream>>>(xa, cD, wvB, cD, bv, nullptr, nullptr, vbb, cD, cD);

    // RoPE -> bf16 q/k
    const size_t nAct = (size_t)cBS * cD;
    rope_kernel<<<(unsigned)((nAct + 255) / 256), 256, 0, stream>>>(qf, kf, qbb, kbb);

    // Attention
    attn_kernel<<<dim3(cS / 64, cH, cB), 128, 0, stream>>>(qbb, kbb, vbb, ob);

    // Output projection + residual
    gemm_bf16_kernel<2><<<gD, 256, 0, stream>>>(ob, cD, woB, cD, bo, xfull, xfull2, nullptr, cD, cD);

    // LN2
    ln_kernel<<<cBS, 256, 0, stream>>>(xfull2, nullptr, ln2_g, ln2_b, nullptr, xfb, 0);

    // FFN: squared-ReLU
    dim3 gF(cDF / 128, cBS / 128);
    gemm_bf16_kernel<3><<<gF, 256, 0, stream>>>(xfb, cD, w1B, cDF, b1, nullptr, nullptr, hb, cD, cDF);
    gemm_bf16_kernel<2><<<gD, 256, 0, stream>>>(hb, cDF, w2B, cD, b2, xfull2, xfull3, nullptr, cDF, cD);

    // Slice last T rows
    const size_t nOut = (size_t)cB * cT * cD;
    extract_kernel<<<(unsigned)((nOut + 255) / 256), 256, 0, stream>>>(xfull3, (float*)d_out);
}
